// MultiHeadAttentionWithDropout_29472065585978
// MI455X (gfx1250) — matrix operand order repaired and checked by reading, not compiled, not run on hardware
//
#include <hip/hip_runtime.h>
#include <hip/hip_bf16.h>
#include <math.h>

// ---------------------------------------------------------------------------
// MultiHeadAttention forward for MI455X (gfx1250, wave32, WMMA 16x16x32 f16).
// B=8, L=1024, D=1024, H=16, dk=64.  ~103 GFLOP, ~0.15 GB HBM -> compute
// bound on the matrix pipes; all GEMM work is v_wmma_f32_16x16x32_f16 with
// f32 accumulation, flash-attention streaming so scores never hit memory.
// ---------------------------------------------------------------------------

#define D_MODEL 1024
#define NHEAD   16
#define DK      64
#define BATCH   8
#define SEQ     1024

typedef _Float16 h4  __attribute__((ext_vector_type(4)));
typedef _Float16 h8  __attribute__((ext_vector_type(8)));
typedef _Float16 h16 __attribute__((ext_vector_type(16)));
typedef float    f4  __attribute__((ext_vector_type(4)));
typedef float    f8  __attribute__((ext_vector_type(8)));

static __device__ __forceinline__ h16 cat16(h8 lo, h8 hi) {
  return __builtin_shufflevector(lo, hi, 0,1,2,3,4,5,6,7,8,9,10,11,12,13,14,15);
}

static __device__ __forceinline__ f8 wmma_16x16x32(h16 a, h16 b, f8 c) {
  // (neg_a, A, neg_b, B, c_mod, C, reuse_a, reuse_b)
  return __builtin_amdgcn_wmma_f32_16x16x32_f16(false, a, false, b, (short)0, c,
                                                false, false);
}

// A fragment (16 rows x 32 K, f16) from a row-major tile.
// ISA layout: lane l holds row l&15; half i holds K=(i>>3)*16+(l>>4)*8+(i&7)
// -> two contiguous 16B chunks at K offsets (l>>4)*8 and 16+(l>>4)*8.
static __device__ __forceinline__ h16 load_frag_a(const _Float16* base,
                                                  int rowStride, int lane) {
  int r = lane & 15, hs = lane >> 4;
  const _Float16* p = base + r * rowStride + hs * 8;
  h8 lo = *(const h8*)p;
  h8 hi = *(const h8*)(p + 16);
  return cat16(lo, hi);
}

// B fragment (32 K x 16 cols, f16). Source stored "column-major": one row of
// `colStride` halves per output column, 32 contiguous K halves inside it.
// ISA layout: half i holds K=(i>>3)*16+(l>>4)*8+(i&7) (same as A).
static __device__ __forceinline__ h16 load_frag_b(const _Float16* base,
                                                  int colStride, int lane) {
  int c = lane & 15, hs = lane >> 4;
  const _Float16* p = base + c * colStride + hs * 8;
  h8 lo = *(const h8*)p;
  h8 hi = *(const h8*)(p + 16);
  return cat16(lo, hi);
}

// ---------------------------------------------------------------------------
// fp32 -> fp16 weight conversion (4 elems / thread)
// ---------------------------------------------------------------------------
__global__ void cvt_f32_f16(const float* __restrict__ in,
                            _Float16* __restrict__ out, int n) {
  int i = (blockIdx.x * blockDim.x + threadIdx.x) * 4;
  if (i + 3 < n) {
    f4 v = *(const f4*)(in + i);
    h4 o = { (_Float16)v.x, (_Float16)v.y, (_Float16)v.z, (_Float16)v.w };
    *(h4*)(out + i) = o;
  }
}

// ---------------------------------------------------------------------------
// Projection GEMM: C[m,n] = A[m,:] . W[n,:] + bias[n]   (x @ W^T + b)
// mode 0: A is fp32 (converted while staging), output f16 head-split
//         out16[((b*H+h)*L + l)*DK + d]  with n = h*64+d, m = b*1024+l
// mode 1: A is f16, output fp32 flat out32[m*D + n]  (final Wo projection)
// Block: 256 thr = 8 waves, tile 128(M) x 64(N), wave tile 32x32 (4 WMMA).
// ---------------------------------------------------------------------------
__global__ __launch_bounds__(256) void proj_gemm(
    const float* __restrict__ A32, const _Float16* __restrict__ A16,
    const _Float16* __restrict__ W, const float* __restrict__ bias,
    _Float16* __restrict__ out16, float* __restrict__ out32, int mode) {
  __shared__ _Float16 As[128 * 40];  // 128 rows x 32 K, stride 40 (16B-aligned rows)
  __shared__ _Float16 Bs[64 * 40];   // 64 cols(n) x 32 K

  const int tid = threadIdx.x;
  const int lane = tid & 31, w = tid >> 5;
  const int wm = w & 3, wn = w >> 2;          // 4x2 wave grid
  const int m0 = blockIdx.x * 128, n0 = blockIdx.y * 64;

  f8 acc[2][2] = {};

  for (int k0 = 0; k0 < D_MODEL; k0 += 32) {
    // ---- stage A (128x32 halves) ----
    for (int c = tid; c < 512; c += 256) {
      int row = c >> 2, k8 = (c & 3) * 8;
      h8 v;
      if (mode == 0) {
        const float* src = A32 + (size_t)(m0 + row) * D_MODEL + k0 + k8;
        f4 x0 = *(const f4*)src;
        f4 x1 = *(const f4*)(src + 4);
        v = h8{ (_Float16)x0.x, (_Float16)x0.y, (_Float16)x0.z, (_Float16)x0.w,
                (_Float16)x1.x, (_Float16)x1.y, (_Float16)x1.z, (_Float16)x1.w };
        if (k0 + 32 < D_MODEL) __builtin_prefetch(src + 32, 0, 1);
      } else {
        const _Float16* src = A16 + (size_t)(m0 + row) * D_MODEL + k0 + k8;
        v = *(const h8*)src;
        if (k0 + 32 < D_MODEL) __builtin_prefetch(src + 32, 0, 1);
      }
      *(h8*)&As[row * 40 + k8] = v;
    }
    // ---- stage B: W rows n0..n0+63, 32 K halves each ----
    {
      int row = tid >> 2, k8 = (tid & 3) * 8;
      const _Float16* src = W + (size_t)(n0 + row) * D_MODEL + k0 + k8;
      *(h8*)&Bs[row * 40 + k8] = *(const h8*)src;
      if (k0 + 32 < D_MODEL) __builtin_prefetch(src + 32, 0, 1);
    }
    __syncthreads();

    h16 a0 = load_frag_a(&As[(wm * 32) * 40], 40, lane);
    h16 a1 = load_frag_a(&As[(wm * 32 + 16) * 40], 40, lane);
    h16 b0 = load_frag_b(&Bs[(wn * 32) * 40], 40, lane);
    h16 b1 = load_frag_b(&Bs[(wn * 32 + 16) * 40], 40, lane);
    acc[0][0] = wmma_16x16x32(a0, b0, acc[0][0]);
    acc[0][1] = wmma_16x16x32(a0, b1, acc[0][1]);
    acc[1][0] = wmma_16x16x32(a1, b0, acc[1][0]);
    acc[1][1] = wmma_16x16x32(a1, b1, acc[1][1]);
    __syncthreads();
  }

  // ---- epilogue: C layout lane=N, vgpr e -> row e + 8*(lane>>4) ----
  for (int ti = 0; ti < 2; ++ti)
    for (int tj = 0; tj < 2; ++tj) {
      int col = n0 + wn * 32 + tj * 16 + (lane & 15);
      float bv = bias[col];
      for (int e = 0; e < 8; ++e) {
        int row = m0 + wm * 32 + ti * 16 + e + 8 * (lane >> 4);
        float val = acc[ti][tj][e] + bv;
        if (mode == 0) {
          int b = row >> 10, l = row & 1023;
          int h = col >> 6, d = col & 63;
          out16[(((size_t)(b * NHEAD + h)) * SEQ + l) * DK + d] = (_Float16)val;
        } else {
          out32[(size_t)row * D_MODEL + col] = val;
        }
      }
    }
}

// ---------------------------------------------------------------------------
// Flash attention: block = one (b,h) x 128 query rows; 8 waves x 16 rows.
// Streams 64-key chunks of K and V^T through LDS; online softmax; P relayout
// C->A goes through wave-private LDS + s_wait_dscnt (LDS is in-order/wave).
// Q pre-scaled by 1/sqrt(dk)=0.125 so S comes out of WMMA already scaled.
// ---------------------------------------------------------------------------
__global__ __launch_bounds__(256) void attn_flash(
    const _Float16* __restrict__ Qh, const _Float16* __restrict__ Kh,
    const _Float16* __restrict__ Vh, const float* __restrict__ amask,
    const unsigned char* __restrict__ kpm, _Float16* __restrict__ outh) {
  __shared__ _Float16 Ks[64 * 72];      // [key][dk]  9.2 KB
  __shared__ _Float16 Vt[64 * 72];      // [dk][key]  9.2 KB (transposed stage)
  __shared__ _Float16 Ps[8][16 * 40];   // per-wave P scratch, 10 KB

  const int tid = threadIdx.x, lane = tid & 31, w = tid >> 5;
  const int bh = blockIdx.x;            // b*NHEAD + h
  const int b = bh >> 4, h = bh & 15;
  const int q0 = blockIdx.y * 128;
  const int wr0 = q0 + w * 16;          // this wave's first query row
  const size_t headoff = (size_t)bh * SEQ * DK;

  // ---- load Q fragments (rows wr0..wr0+15, dk 0..63), scale by 0.125 ----
  h16 qa0, qa1;
  {
    int r = lane & 15, hs = lane >> 4;
    const _Float16* qp = Qh + headoff + (size_t)(wr0 + r) * DK + hs * 8;
    h8 v0 = *(const h8*)qp;          // dk (l>>4)*8 .. +7
    h8 v1 = *(const h8*)(qp + 16);   // dk 16+(l>>4)*8 ..
    h8 v2 = *(const h8*)(qp + 32);
    h8 v3 = *(const h8*)(qp + 48);
    qa0 = cat16(v0, v1);
    qa1 = cat16(v2, v3);
    for (int i = 0; i < 16; ++i) {
      qa0[i] *= (_Float16)0.125f;
      qa1[i] *= (_Float16)0.125f;
    }
  }

  f8 o[4] = {};                         // 16 rows x 64 dk output accumulator
  float m_[8], lsum[8];
  for (int e = 0; e < 8; ++e) { m_[e] = -3.0e38f; lsum[e] = 0.0f; }

  for (int kk0 = 0; kk0 < SEQ; kk0 += 64) {
    // ---- stage K chunk and transposed V chunk ----
    for (int c = tid; c < 512; c += 256) {
      int key = c >> 3, d8 = (c & 7) * 8;
      const _Float16* kp = Kh + headoff + (size_t)(kk0 + key) * DK + d8;
      const _Float16* vp = Vh + headoff + (size_t)(kk0 + key) * DK + d8;
      h8 kv = *(const h8*)kp;
      *(h8*)&Ks[key * 72 + d8] = kv;
      h8 vv = *(const h8*)vp;
      for (int i = 0; i < 8; ++i) Vt[(d8 + i) * 72 + key] = vv[i];
      if (kk0 + 64 < SEQ) {
        __builtin_prefetch(kp + 64 * DK, 0, 1);
        __builtin_prefetch(vp + 64 * DK, 0, 1);
      }
    }
    __syncthreads();

    for (int kg = 0; kg < 2; ++kg) {    // two 32-key groups
      // ---- scores: two 16-key tiles, each accumulated over dk in 2 WMMAs ----
      f8 s[2];
      for (int t = 0; t < 2; ++t) {
        int keyl = kg * 32 + t * 16;
        h16 bk0 = load_frag_b(&Ks[keyl * 72], 72, lane);       // dk 0..31
        h16 bk1 = load_frag_b(&Ks[keyl * 72 + 32], 72, lane);  // dk 32..63
        f8 z = {};
        z = wmma_16x16x32(qa0, bk0, z);
        z = wmma_16x16x32(qa1, bk1, z);
        // additive attn_mask + key_padding_mask (faithful to reference)
        int colg = kk0 + keyl + (lane & 15);
        unsigned char kp = kpm[b * SEQ + colg];
        const float* mrow =
            amask + (size_t)(wr0 + 8 * (lane >> 4)) * SEQ + colg;
        for (int e = 0; e < 8; ++e) {
          float sv = z[e] + mrow[(size_t)e * SEQ];
          if (kp) sv = -INFINITY;
          z[e] = sv;
        }
        s[t] = z;
      }

      // ---- online softmax over these 32 columns (reduce in 16-lane halves) --
      float alpha[8];
      for (int e = 0; e < 8; ++e) {
        float rv = fmaxf(s[0][e], s[1][e]);
        rv = fmaxf(rv, __shfl_xor(rv, 1));
        rv = fmaxf(rv, __shfl_xor(rv, 2));
        rv = fmaxf(rv, __shfl_xor(rv, 4));
        rv = fmaxf(rv, __shfl_xor(rv, 8));
        float mn = fmaxf(m_[e], rv);
        float al = __expf(m_[e] - mn);
        float p0 = __expf(s[0][e] - mn);
        float p1 = __expf(s[1][e] - mn);
        s[0][e] = p0; s[1][e] = p1;
        float rs = p0 + p1;
        rs += __shfl_xor(rs, 1);
        rs += __shfl_xor(rs, 2);
        rs += __shfl_xor(rs, 4);
        rs += __shfl_xor(rs, 8);
        lsum[e] = lsum[e] * al + rs;
        m_[e] = mn;
        alpha[e] = al;
      }
      for (int j = 0; j < 4; ++j)
        for (int e = 0; e < 8; ++e) o[j][e] *= alpha[e];

      // ---- P (C layout) -> f16 A layout via wave-private LDS scratch ----
      {
        _Float16* ps = &Ps[w][0];
        int r8 = 8 * (lane >> 4), c0 = lane & 15;
        for (int t = 0; t < 2; ++t)
          for (int e = 0; e < 8; ++e)
            ps[(r8 + e) * 40 + t * 16 + c0] = (_Float16)s[t][e];
      }
      // LDS is in-order per wave; this blocks compiler reordering + drains DS
      asm volatile("s_wait_dscnt 0" ::: "memory");
      h16 pa = load_frag_a(&Ps[w][0], 40, lane);

      // ---- O += P @ V : 4 dk tiles of 16 ----
      for (int j = 0; j < 4; ++j) {
        h16 bv = load_frag_b(&Vt[(j * 16) * 72 + kg * 32], 72, lane);
        o[j] = wmma_16x16x32(pa, bv, o[j]);
      }
    }
    __syncthreads();
  }

  // ---- normalize and store concat-head f16 [B, L, D] ----
  for (int e = 0; e < 8; ++e) lsum[e] = 1.0f / lsum[e];
  {
    int r8 = 8 * (lane >> 4), c0 = lane & 15;
    for (int j = 0; j < 4; ++j)
      for (int e = 0; e < 8; ++e) {
        int row = wr0 + r8 + e;
        int col = h * DK + j * 16 + c0;
        outh[(size_t)(b * SEQ + row) * D_MODEL + col] =
            (_Float16)(o[j][e] * lsum[e]);
      }
  }
}

// ---------------------------------------------------------------------------
// Host-side launch
// ---------------------------------------------------------------------------
extern "C" void kernel_launch(void* const* d_in, const int* in_sizes, int n_in,
                              void* d_out, int out_size, void* d_ws,
                              size_t ws_size, hipStream_t stream) {
  (void)in_sizes; (void)n_in; (void)out_size; (void)ws_size;
  const float* query = (const float*)d_in[0];
  const float* key_  = (const float*)d_in[1];
  const float* value = (const float*)d_in[2];
  const float* amask = (const float*)d_in[3];
  const unsigned char* kpm = (const unsigned char*)d_in[4];
  const float* Wq = (const float*)d_in[5];
  const float* bq = (const float*)d_in[6];
  const float* Wk = (const float*)d_in[7];
  const float* bk = (const float*)d_in[8];
  const float* Wv = (const float*)d_in[9];
  const float* bv = (const float*)d_in[10];
  const float* Wo = (const float*)d_in[11];
  const float* bo = (const float*)d_in[12];
  float* out = (float*)d_out;

  // workspace carve-up (all f16): 4 weights (8MB) + Qh/Kh/Vh (48MB) + AO (16MB)
  const size_t WELEM = (size_t)D_MODEL * D_MODEL;        // 1M elems
  const size_t XELEM = (size_t)BATCH * SEQ * D_MODEL;    // 8M elems
  _Float16* Wq16 = (_Float16*)d_ws;
  _Float16* Wk16 = Wq16 + WELEM;
  _Float16* Wv16 = Wk16 + WELEM;
  _Float16* Wo16 = Wv16 + WELEM;
  _Float16* Qh   = Wo16 + WELEM;   // [B,H,L,DK]
  _Float16* Kh16 = Qh + XELEM;
  _Float16* Vh16 = Kh16 + XELEM;
  _Float16* AO   = Vh16 + XELEM;   // attention out, [B,L,D] f16

  dim3 blk(256);
  dim3 gcvt((unsigned)(WELEM / 1024));
  cvt_f32_f16<<<gcvt, blk, 0, stream>>>(Wq, Wq16, (int)WELEM);
  cvt_f32_f16<<<gcvt, blk, 0, stream>>>(Wk, Wk16, (int)WELEM);
  cvt_f32_f16<<<gcvt, blk, 0, stream>>>(Wv, Wv16, (int)WELEM);
  cvt_f32_f16<<<gcvt, blk, 0, stream>>>(Wo, Wo16, (int)WELEM);

  dim3 ggemm(BATCH * SEQ / 128, D_MODEL / 64);  // (64, 16)
  proj_gemm<<<ggemm, blk, 0, stream>>>(query, nullptr, Wq16, bq, Qh, nullptr, 0);
  proj_gemm<<<ggemm, blk, 0, stream>>>(key_,  nullptr, Wk16, bk, Kh16, nullptr, 0);
  proj_gemm<<<ggemm, blk, 0, stream>>>(value, nullptr, Wv16, bv, Vh16, nullptr, 0);

  attn_flash<<<dim3(BATCH * NHEAD, SEQ / 128), blk, 0, stream>>>(
      Qh, Kh16, Vh16, amask, kpm, AO);

  proj_gemm<<<ggemm, blk, 0, stream>>>(nullptr, AO, Wo16, bo, nullptr, out, 1);
}
